// AttentiveWeightedGIN_46626164965922
// MI455X (gfx1250) — matrix operand order repaired and checked by reading, not compiled, not run on hardware
//
#include <hip/hip_runtime.h>
#include <hip/hip_bf16.h>

// ---------------- problem constants (match reference) ----------------
#define NN 50000      // nodes
#define EE 800000     // edges (before self loops)
#define E2 850000     // edges + self loops
#define VV 30000      // vocab
#define DD 128        // emb/hidden dim
#define GG 64         // graphs
#define LL 2          // layers
#define NEG_SLOPE 0.2f
#define LN_EPS 1e-5f

typedef _Float16 v2h  __attribute__((ext_vector_type(2)));
typedef _Float16 v4h  __attribute__((ext_vector_type(4)));
typedef _Float16 v8h  __attribute__((ext_vector_type(8)));
typedef _Float16 v16h __attribute__((ext_vector_type(16)));
typedef float    v8f  __attribute__((ext_vector_type(8)));

union AFrag { v16h v; v8h h[2]; };

// ------------- ordered-uint encoding for float atomic max -------------
__device__ __forceinline__ unsigned fenc(float f) {
    unsigned u = __float_as_uint(f);
    return (u & 0x80000000u) ? ~u : (u | 0x80000000u);
}
__device__ __forceinline__ float fdec(unsigned u) {
    return (u & 0x80000000u) ? __uint_as_float(u ^ 0x80000000u)
                             : __uint_as_float(~u);
}

// ---------------------------------------------------------------------
// 1) Embedding gather: x[n,:] = emb[node_ids[n],:]
// ---------------------------------------------------------------------
__global__ void k_gather(const int* __restrict__ node_ids,
                         const float* __restrict__ emb,
                         float* __restrict__ x) {
    int idx = blockIdx.x * blockDim.x + threadIdx.x;   // N*32 threads, float4 each
    int n = idx >> 5;
    int c = idx & 31;
    if (n >= NN) return;
    int vid = node_ids[n];
    ((float4*)x)[n * 32 + c] = ((const float4*)emb)[vid * 32 + c];
}

// ---------------------------------------------------------------------
// 2) Per-layer reset: agg=0, denom=0, smax=enc(-inf)~0
// ---------------------------------------------------------------------
__global__ void k_reset(float* __restrict__ agg,
                        float* __restrict__ denom,
                        unsigned* __restrict__ smax) {
    int idx = blockIdx.x * blockDim.x + threadIdx.x;   // N*128 threads
    if (idx < NN * DD) agg[idx] = 0.0f;
    if (idx < NN) { denom[idx] = 0.0f; smax[idx] = 0u; }
}

// ---------------------------------------------------------------------
// 3) LayerNorm + per-node attention partial dots (wave per node, wave32)
//    xn = LN(x)*g+b ; sd[n]=xn.aw[:D] ; ss[n]=xn.aw[D:]
// ---------------------------------------------------------------------
__global__ void k_ln_score(const float* __restrict__ x,
                           float* __restrict__ xn,
                           float* __restrict__ sd,
                           float* __restrict__ ss,
                           const float* __restrict__ g,
                           const float* __restrict__ b,
                           const float* __restrict__ aw) {
    int lane = threadIdx.x & 31;
    int n = blockIdx.x * 8 + (threadIdx.x >> 5);
    if (n >= NN) return;
    const float* row = x + (long)n * DD + lane * 4;
    float v0 = row[0], v1 = row[1], v2 = row[2], v3 = row[3];
    float s = v0 + v1 + v2 + v3;
    #pragma unroll
    for (int o = 16; o > 0; o >>= 1) s += __shfl_xor(s, o, 32);
    float mu = s * (1.0f / DD);
    float d0 = v0 - mu, d1 = v1 - mu, d2 = v2 - mu, d3 = v3 - mu;
    float q = d0 * d0 + d1 * d1 + d2 * d2 + d3 * d3;
    #pragma unroll
    for (int o = 16; o > 0; o >>= 1) q += __shfl_xor(q, o, 32);
    float rs = rsqrtf(q * (1.0f / DD) + LN_EPS);
    int c = lane * 4;
    float y0 = d0 * rs * g[c + 0] + b[c + 0];
    float y1 = d1 * rs * g[c + 1] + b[c + 1];
    float y2 = d2 * rs * g[c + 2] + b[c + 2];
    float y3 = d3 * rs * g[c + 3] + b[c + 3];
    float* orow = xn + (long)n * DD + c;
    orow[0] = y0; orow[1] = y1; orow[2] = y2; orow[3] = y3;
    // attention partials: dst part aw[0:128], src part aw[128:256]
    float pd = y0 * aw[c] + y1 * aw[c + 1] + y2 * aw[c + 2] + y3 * aw[c + 3];
    float ps = y0 * aw[DD + c] + y1 * aw[DD + c + 1] + y2 * aw[DD + c + 2] + y3 * aw[DD + c + 3];
    #pragma unroll
    for (int o = 16; o > 0; o >>= 1) { pd += __shfl_xor(pd, o, 32); ps += __shfl_xor(ps, o, 32); }
    if (lane == 0) { sd[n] = pd; ss[n] = ps; }
}

// ---------------------------------------------------------------------
// 4) Edge pass 1: score + segment max
// ---------------------------------------------------------------------
__global__ void k_edge_max(const int* __restrict__ ei,
                           const float* __restrict__ sd,
                           const float* __restrict__ ss,
                           float* __restrict__ score,
                           unsigned* __restrict__ smax) {
    int e = blockIdx.x * blockDim.x + threadIdx.x;
    if (e >= E2) return;
    int src = (e < EE) ? ei[e]      : (e - EE);
    int dst = (e < EE) ? ei[EE + e] : (e - EE);
    float s = sd[dst] + ss[src];
    s = (s >= 0.0f) ? s : NEG_SLOPE * s;
    score[e] = s;
    atomicMax(&smax[dst], fenc(s));
}

// ---------------------------------------------------------------------
// 5) Edge pass 2: exp + segment sum
// ---------------------------------------------------------------------
__global__ void k_edge_exp(const int* __restrict__ ei,
                           const float* __restrict__ score,
                           const unsigned* __restrict__ smax,
                           float* __restrict__ exs,
                           float* __restrict__ denom) {
    int e = blockIdx.x * blockDim.x + threadIdx.x;
    if (e >= E2) return;
    int dst = (e < EE) ? ei[EE + e] : (e - EE);
    float ex = __expf(score[e] - fdec(smax[dst]));
    exs[e] = ex;
    atomicAdd(&denom[dst], ex);
}

// ---------------------------------------------------------------------
// 6) Edge pass 3: alpha*x[src] scatter-add (one wave per edge)
// ---------------------------------------------------------------------
__global__ void k_edge_scatter(const int* __restrict__ ei,
                               const float* __restrict__ ew,
                               const float* __restrict__ exs,
                               const float* __restrict__ denom,
                               const float* __restrict__ xn,
                               float* __restrict__ agg) {
    int idx = blockIdx.x * blockDim.x + threadIdx.x;   // E2*32 threads
    int e = idx >> 5;
    int lane = idx & 31;
    if (e >= E2) return;
    int src = (e < EE) ? ei[e]      : (e - EE);
    int dst = (e < EE) ? ei[EE + e] : (e - EE);
    // prefetch the source row while scalar work resolves (global_prefetch_b8)
    __builtin_prefetch(xn + (long)src * DD + lane * 4, 0, 1);
    float w = (e < EE) ? ew[e] : 1.0f;
    float alpha = exs[e] / (denom[dst] + 1e-16f) * w;
    int c = lane * 4;
    float4 xv = ((const float4*)(xn + (long)src * DD))[lane];
    float* ap = agg + (long)dst * DD + c;
    atomicAdd(ap + 0, alpha * xv.x);
    atomicAdd(ap + 1, alpha * xv.y);
    atomicAdd(ap + 2, alpha * xv.z);
    atomicAdd(ap + 3, alpha * xv.w);
}

// ---------------------------------------------------------------------
// 7) GIN update + fused 2-matmul MLP via v_wmma_f32_16x16x32_f16
//    block = 256 threads (8 waves), owns 128 rows (nodes).
//    LDS: A(f16 128x128, [m][k]) | W1^T(f16 [n][k]) | W2^T(f16 [n][k]) | H([m][n])
//
//    Both stages are computed TRANSPOSED:
//      stage1: H^T = W1^T @ A^T   (A-frag from aW1, B-frag from aA)
//      stage2: out^T = W2^T @ H^T (A-frag from aW2, B-frag from aH)
//    With the C/D layout (lane = column = node row, 8 consecutive rows = 8
//    consecutive n per lane), epilogues become contiguous:
//      H write-back  -> one ds_store_b128 per tile per lane
//      global output -> two global_store_b128 per tile per lane
//    All fragment loads remain contiguous ds_load_b128.
// ---------------------------------------------------------------------
__global__ void k_gin_mlp(const float* __restrict__ xn,
                          const float* __restrict__ agg,
                          const float* __restrict__ W1,
                          const float* __restrict__ B1,
                          const float* __restrict__ W2,
                          const float* __restrict__ B2,
                          const float* __restrict__ epsp,
                          float* __restrict__ xout) {
    extern __shared__ _Float16 smem[];
    _Float16* aA  = smem;                 // A  [m][k]  128x128
    _Float16* aW1 = smem + 16384;         // W1^T [n][k]
    _Float16* aW2 = smem + 2 * 16384;     // W2^T [n][k]
    _Float16* aH  = smem + 3 * 16384;     // H  [m][n]

    int tid = threadIdx.x;
    int rowBase = blockIdx.x * 128;
    float ep1 = 1.0f + epsp[0];

    // stage weights transposed (W[k][n] -> LDS[n*128+k]); float4 global reads
    for (int i = tid; i < 4096; i += 256) {
        int k = i >> 5;             // weight row
        int n4 = (i & 31) * 4;      // 4 consecutive cols
        float4 w1v = ((const float4*)W1)[i];
        float4 w2v = ((const float4*)W2)[i];
        aW1[(n4 + 0) * 128 + k] = (_Float16)w1v.x;
        aW1[(n4 + 1) * 128 + k] = (_Float16)w1v.y;
        aW1[(n4 + 2) * 128 + k] = (_Float16)w1v.z;
        aW1[(n4 + 3) * 128 + k] = (_Float16)w1v.w;
        aW2[(n4 + 0) * 128 + k] = (_Float16)w2v.x;
        aW2[(n4 + 1) * 128 + k] = (_Float16)w2v.y;
        aW2[(n4 + 2) * 128 + k] = (_Float16)w2v.z;
        aW2[(n4 + 3) * 128 + k] = (_Float16)w2v.w;
    }
    // stage A block: (1+eps)*xn + agg, packed f16, 8B LDS stores
    for (int i = tid; i < 4096; i += 256) {
        int r = i >> 5;             // local row
        int c4 = (i & 31) * 4;
        int gr = rowBase + r;
        v4h pk;
        if (gr < NN) {
            long o = ((long)gr * DD + c4) >> 2;
            float4 xv = ((const float4*)xn)[o];
            float4 av = ((const float4*)agg)[o];
            pk[0] = (_Float16)(ep1 * xv.x + av.x);
            pk[1] = (_Float16)(ep1 * xv.y + av.y);
            pk[2] = (_Float16)(ep1 * xv.z + av.z);
            pk[3] = (_Float16)(ep1 * xv.w + av.w);
        } else {
            pk[0] = pk[1] = pk[2] = pk[3] = (_Float16)0.0f;
        }
        *(v4h*)&aA[r * 128 + c4] = pk;
    }
    __syncthreads();

    int lane = tid & 31;
    int lm = lane & 15;
    int hi = lane >> 4;            // 0 or 1
    int khi = hi * 16;             // A-layout K offset for hi half-wave
    int n0 = (tid >> 5) * 16;      // this wave's 16 output-feature rows

    // ============ stage 1 (transposed): H^T = W1^T @ A^T ============
    AFrag afr[4];
    #pragma unroll
    for (int kc = 0; kc < 4; ++kc) {
        const _Float16* ap = &aW1[(n0 + lm) * 128 + kc * 32 + khi];
        afr[kc].h[0] = *(const v8h*)(ap);
        afr[kc].h[1] = *(const v8h*)(ap + 8);
    }
    float bv1[8];
    #pragma unroll
    for (int r = 0; r < 8; ++r) bv1[r] = B1[n0 + r + hi * 8];

    #pragma unroll
    for (int nt = 0; nt < 8; ++nt) {
        int m = nt * 16 + lm;          // local node row (B-frag column)
        const _Float16* bp = &aA[m * 128 + hi * 16];
        v16h b0 = *(const v16h*)(bp + 0);
        v16h b1 = *(const v16h*)(bp + 32);
        v16h b2 = *(const v16h*)(bp + 64);
        v16h b3 = *(const v16h*)(bp + 96);
        v8f acc0, acc1;
        #pragma unroll
        for (int i = 0; i < 8; ++i) { acc0[i] = bv1[i]; acc1[i] = 0.0f; }
        acc0 = __builtin_amdgcn_wmma_f32_16x16x32_f16(false, afr[0].v, false, b0, (short)0, acc0, false, false);
        acc1 = __builtin_amdgcn_wmma_f32_16x16x32_f16(false, afr[1].v, false, b1, (short)0, acc1, false, false);
        acc0 = __builtin_amdgcn_wmma_f32_16x16x32_f16(false, afr[2].v, false, b2, (short)0, acc0, false, false);
        acc1 = __builtin_amdgcn_wmma_f32_16x16x32_f16(false, afr[3].v, false, b3, (short)0, acc1, false, false);
        // lane holds node row m, 8 consecutive n = n0+hi*8+r  ->  16B store
        v8h hv;
        #pragma unroll
        for (int r = 0; r < 8; ++r) {
            float v = acc0[r] + acc1[r];
            hv[r] = (_Float16)(v > 0.0f ? v : 0.0f);
        }
        *(v8h*)&aH[m * 128 + n0 + hi * 8] = hv;
    }
    __syncthreads();

    // ============ stage 2 (transposed): out^T = W2^T @ H^T ============
    #pragma unroll
    for (int kc = 0; kc < 4; ++kc) {
        const _Float16* ap = &aW2[(n0 + lm) * 128 + kc * 32 + khi];
        afr[kc].h[0] = *(const v8h*)(ap);
        afr[kc].h[1] = *(const v8h*)(ap + 8);
    }
    float bv2[8];
    #pragma unroll
    for (int r = 0; r < 8; ++r) bv2[r] = B2[n0 + r + hi * 8];

    #pragma unroll
    for (int nt = 0; nt < 8; ++nt) {
        int m = nt * 16 + lm;          // local node row
        const _Float16* bp = &aH[m * 128 + hi * 16];
        v16h b0 = *(const v16h*)(bp + 0);
        v16h b1 = *(const v16h*)(bp + 32);
        v16h b2 = *(const v16h*)(bp + 64);
        v16h b3 = *(const v16h*)(bp + 96);
        v8f acc0, acc1;
        #pragma unroll
        for (int i = 0; i < 8; ++i) { acc0[i] = bv2[i]; acc1[i] = 0.0f; }
        acc0 = __builtin_amdgcn_wmma_f32_16x16x32_f16(false, afr[0].v, false, b0, (short)0, acc0, false, false);
        acc1 = __builtin_amdgcn_wmma_f32_16x16x32_f16(false, afr[1].v, false, b1, (short)0, acc1, false, false);
        acc0 = __builtin_amdgcn_wmma_f32_16x16x32_f16(false, afr[2].v, false, b2, (short)0, acc0, false, false);
        acc1 = __builtin_amdgcn_wmma_f32_16x16x32_f16(false, afr[3].v, false, b3, (short)0, acc1, false, false);
        int gr = rowBase + m;
        if (gr < NN) {
            float4 f0, f1;
            float v;
            v = acc0[0] + acc1[0]; f0.x = v > 0.0f ? v : 0.0f;
            v = acc0[1] + acc1[1]; f0.y = v > 0.0f ? v : 0.0f;
            v = acc0[2] + acc1[2]; f0.z = v > 0.0f ? v : 0.0f;
            v = acc0[3] + acc1[3]; f0.w = v > 0.0f ? v : 0.0f;
            v = acc0[4] + acc1[4]; f1.x = v > 0.0f ? v : 0.0f;
            v = acc0[5] + acc1[5]; f1.y = v > 0.0f ? v : 0.0f;
            v = acc0[6] + acc1[6]; f1.z = v > 0.0f ? v : 0.0f;
            v = acc0[7] + acc1[7]; f1.w = v > 0.0f ? v : 0.0f;
            float* op = xout + (long)gr * DD + n0 + hi * 8;
            ((float4*)op)[0] = f0;
            ((float4*)op)[1] = f1;
        }
    }
}

// ---------------------------------------------------------------------
// 8) pooling
// ---------------------------------------------------------------------
__global__ void k_reset_pool(float* __restrict__ gsum, float* __restrict__ cnt) {
    int i = threadIdx.x;
    for (int j = i; j < GG * DD; j += 256) gsum[j] = 0.0f;
    if (i < GG) cnt[i] = 0.0f;
}

__global__ void k_pool(const int* __restrict__ batch,
                       const float* __restrict__ x,
                       float* __restrict__ gsum,
                       float* __restrict__ cnt) {
    int idx = blockIdx.x * blockDim.x + threadIdx.x;   // N*32 threads
    int n = idx >> 5;
    int lane = idx & 31;
    if (n >= NN) return;
    int b = batch[n];
    float4 xv = ((const float4*)(x + (long)n * DD))[lane];
    float* gp = gsum + b * DD + lane * 4;
    atomicAdd(gp + 0, xv.x);
    atomicAdd(gp + 1, xv.y);
    atomicAdd(gp + 2, xv.z);
    atomicAdd(gp + 3, xv.w);
    if (lane == 0) atomicAdd(&cnt[b], 1.0f);
}

// ---------------------------------------------------------------------
// 9) head MLP: out = relu(g@wh1+bh1)@wh2+bh2  ->  [64,2]
// ---------------------------------------------------------------------
__global__ void k_head(const float* __restrict__ gsum,
                       const float* __restrict__ cnt,
                       const float* __restrict__ wh1,
                       const float* __restrict__ bh1,
                       const float* __restrict__ wh2,
                       const float* __restrict__ bh2,
                       float* __restrict__ out) {
    __shared__ float gm[GG * DD];       // 32 KB
    __shared__ float hg[GG * 64];       // 16 KB
    int tid = threadIdx.x;
    for (int i = tid; i < GG * DD; i += 256) {
        int gi = i >> 7;
        gm[i] = gsum[i] / fmaxf(cnt[gi], 1.0f);
    }
    __syncthreads();
    for (int i = tid; i < GG * 64; i += 256) {
        int gi = i >> 6, j = i & 63;
        float acc = bh1[j];
        #pragma unroll 8
        for (int k = 0; k < DD; ++k) acc += gm[gi * DD + k] * wh1[k * 64 + j];
        hg[i] = acc > 0.0f ? acc : 0.0f;
    }
    __syncthreads();
    for (int i = tid; i < GG * 2; i += 256) {
        int gi = i >> 1, j = i & 1;
        float acc = bh2[j];
        #pragma unroll
        for (int k = 0; k < 64; ++k) acc += hg[gi * 64 + k] * wh2[k * 2 + j];
        out[i] = acc;
    }
}

// ---------------------------------------------------------------------
extern "C" void kernel_launch(void* const* d_in, const int* in_sizes, int n_in,
                              void* d_out, int out_size, void* d_ws, size_t ws_size,
                              hipStream_t stream) {
    const int*   node_ids = (const int*)  d_in[0];
    const int*   ei       = (const int*)  d_in[1];   // [2,E] row-major
    const float* ew       = (const float*)d_in[2];
    const int*   batch    = (const int*)  d_in[3];
    const float* emb      = (const float*)d_in[4];
    const float* ln_g     = (const float*)d_in[5];   // [L,D]
    const float* ln_b     = (const float*)d_in[6];
    const float* att_w    = (const float*)d_in[7];   // [L,2D]
    const float* w1       = (const float*)d_in[8];   // [L,D,D]
    const float* b1       = (const float*)d_in[9];
    const float* w2       = (const float*)d_in[10];
    const float* b2       = (const float*)d_in[11];
    const float* epsv     = (const float*)d_in[12];  // [L]
    const float* wh1      = (const float*)d_in[13];
    const float* bh1      = (const float*)d_in[14];
    const float* wh2      = (const float*)d_in[15];
    const float* bh2      = (const float*)d_in[16];
    float* out = (float*)d_out;

    // workspace bump allocator (16B aligned)
    char* ws = (char*)d_ws;
    size_t off = 0;
    auto alloc = [&](size_t bytes) -> char* {
        char* p = ws + off;
        off += (bytes + 15) & ~((size_t)15);
        return p;
    };
    float*    x     = (float*)   alloc((size_t)NN * DD * 4);
    float*    xn    = (float*)   alloc((size_t)NN * DD * 4);
    float*    agg   = (float*)   alloc((size_t)NN * DD * 4);
    float*    sd    = (float*)   alloc((size_t)NN * 4);
    float*    ss    = (float*)   alloc((size_t)NN * 4);
    float*    denom = (float*)   alloc((size_t)NN * 4);
    unsigned* smax  = (unsigned*)alloc((size_t)NN * 4);
    float*    score = (float*)   alloc((size_t)E2 * 4);
    float*    exs   = (float*)   alloc((size_t)E2 * 4);
    float*    gsum  = (float*)   alloc((size_t)GG * DD * 4);
    float*    cnt   = (float*)   alloc((size_t)GG * 4);
    (void)ws_size; (void)n_in; (void)in_sizes; (void)out_size;

    const int T = 256;
    k_gather<<<(NN * 32 + T - 1) / T, T, 0, stream>>>(node_ids, emb, x);

    for (int l = 0; l < LL; ++l) {
        k_reset<<<(NN * DD + T - 1) / T, T, 0, stream>>>(agg, denom, smax);
        k_ln_score<<<(NN + 7) / 8, T, 0, stream>>>(
            x, xn, sd, ss, ln_g + l * DD, ln_b + l * DD, att_w + l * 2 * DD);
        k_edge_max<<<(E2 + T - 1) / T, T, 0, stream>>>(ei, sd, ss, score, smax);
        k_edge_exp<<<(E2 + T - 1) / T, T, 0, stream>>>(ei, score, smax, exs, denom);
        k_edge_scatter<<<((size_t)E2 * 32 + T - 1) / T, T, 0, stream>>>(
            ei, ew, exs, denom, xn, agg);
        k_gin_mlp<<<(NN + 127) / 128, T, 4 * 16384 * sizeof(_Float16), stream>>>(
            xn, agg, w1 + (size_t)l * DD * DD, b1 + l * DD,
            w2 + (size_t)l * DD * DD, b2 + l * DD, epsv + l, x);
    }

    k_reset_pool<<<1, T, 0, stream>>>(gsum, cnt);
    k_pool<<<(NN * 32 + T - 1) / T, T, 0, stream>>>(batch, x, gsum, cnt);
    k_head<<<1, T, 0, stream>>>(gsum, cnt, wh1, bh1, wh2, bh2, out);
}
